// FEFM_35218731827351
// MI455X (gfx1250) — matrix operand order repaired and checked by reading, not compiled, not run on hardware
//
#include <hip/hip_runtime.h>

// ---------------- problem constants ----------------
#define Bn 8
#define Cc 96
#define Hh 128
#define Ww 128
#define HW (Hh*Ww)
static constexpr size_t TS = (size_t)Bn * Cc * HW;   // 12,582,912 floats per tensor

typedef __attribute__((ext_vector_type(16))) __bf16       v16bf;
typedef __attribute__((ext_vector_type(8)))  float        v8f;
typedef __attribute__((ext_vector_type(8)))  unsigned int v8ui;

// DCT-II (unnormalized, torch_dct style): D[k][n] = 2*cos(pi*(2n+1)k/8)
__constant__ float c_D[16] = {
    2.0f,                2.0f,                2.0f,                2.0f,
    1.8477590650225735f, 0.7653668647301796f,-0.7653668647301796f,-1.8477590650225735f,
    1.4142135623730951f,-1.4142135623730951f,-1.4142135623730951f, 1.4142135623730951f,
    0.7653668647301796f,-1.8477590650225735f, 1.8477590650225735f,-0.7653668647301796f };

// Fragment index helpers (ISA 7.12.2 layouts, wave32).
// A (16x32 bf16): lane m = lane&15, h = lane>>4; VGPR v holds K pair:
__device__ __forceinline__ int a_kb(int v, int h) {
    return ((v & 4) ? 16 : 0) + h * 8 + (v & 3) * 2;
}
// B (32x16 bf16): lane n = lane&15, h = lane>>4; VGPR v holds same K pair as A
__device__ __forceinline__ int b_kb(int v, int h) { return ((v & 4) ? 16 : 0) + h * 8 + (v & 3) * 2; }
// C/D (16x16 f32): lane n = lane&15; VGPR v holds row m = v + 8*(lane>>4)

__device__ __forceinline__ v8f wmma_bf16(const v16bf& a, const v16bf& b, const v8f& c) {
    return __builtin_amdgcn_wmma_f32_16x16x32_bf16(false, a, false, b, (short)0, c,
                                                   false, false);
}

// ============================================================================
// 1) conv3x3 as implicit GEMM: M = co (6 tiles of 16), K = 9 taps * 96 ci
//    (27 chunks of 32), N = spatial (B*H*W / 16 tiles). One wave per 16x16
//    output tile. All 8 waves of a block share one co-tile, so the 16x864
//    weight slab is staged once into LDS as packed bf16 pairs; A fragments
//    become single ds_load_b32 reads. Halo handling: clamped (always valid)
//    address, fused v_cvt_pk_bf16_f32 conversion, then one cndmask_b32 per
//    packed pair.
// ============================================================================
__global__ void __launch_bounds__(256)
conv3x3_wmma(const float* __restrict__ x, const float* __restrict__ w,
             const float* __restrict__ bias, float* __restrict__ y)
{
    // bf16 pair slab: [tap(9)][coL(16)][ciPair(48)] -> 6912 dwords = 27.6 KB
    __shared__ unsigned int lds_w[9 * 16 * 48];

    const int tid  = threadIdx.x;
    const int wid0 = blockIdx.x * 8;
    const int coT  = wid0 >> 13;                 // uniform across the block

    // ---- stage weight tile (16 co x 96 ci x 9 taps) into LDS as bf16 ----
    for (int p = tid; p < 9 * 16 * 48; p += 256) {
        const int tap = p / 768;
        const int rp  = p - tap * 768;
        const int coL = rp / 48;
        const int ci  = (rp - coL * 48) * 2;
        const float* wp = w + (size_t)(coT * 16 + coL) * (Cc * 9) + tap;
        union { __bf16 h2[2]; unsigned int u; } pk;
        pk.h2[0] = (__bf16)wp[ci * 9];
        pk.h2[1] = (__bf16)wp[(ci + 1) * 9];
        lds_w[p] = pk.u;
    }
    __syncthreads();

    const int lane = tid & 31;
    const int wid  = wid0 + (tid >> 5);          // 0..49151
    const int sT   = wid & 8191;
    const int s0   = sT * 16;
    const int b    = s0 / HW;
    const int rem  = s0 - b * HW;
    const int y0   = rem >> 7;                   // row (W=128)
    const int x0   = rem & 127;                  // col base (multiple of 16)

    const int m15 = lane & 15;
    const int h   = lane >> 4;
    const int n   = m15;                         // output column (spatial)

    v8f acc;
#pragma unroll
    for (int v = 0; v < 8; ++v) acc[v] = bias[coT * 16 + v + 8 * h];

    const float* xb = x + (size_t)b * Cc * HW;

    for (int tap = 0; tap < 9; ++tap) {
        const int ty = tap / 3, tx = tap - ty * 3;
        const int yy = y0 + ty - 1;
        const int xx = x0 + n + tx - 1;
        const bool ok  = ((unsigned)yy < 128u) & ((unsigned)xx < 128u);
        const int  yyc = yy < 0 ? 0 : (yy > 127 ? 127 : yy);
        const int  xxc = xx < 0 ? 0 : (xx > 127 ? 127 : xx);
        const float* xrow = xb + yyc * Ww + xxc;       // always in-bounds
        const unsigned int* lw = lds_w + tap * 768 + m15 * 48;
#pragma unroll
        for (int cc = 0; cc < 3; ++cc) {
            const int cib = cc * 32;
            union { v16bf bf; v8ui u; } af, bfr;
#pragma unroll
            for (int v = 0; v < 8; ++v) {
                af.u[v] = lw[(cib + a_kb(v, h)) >> 1];     // packed bf16 pair
                const int cj = cib + b_kb(v, h);
                bfr.bf[2 * v]     = (__bf16)xrow[(size_t)cj * HW];       // fused
                bfr.bf[2 * v + 1] = (__bf16)xrow[(size_t)(cj + 1) * HW]; // cvt_pk
            }
#pragma unroll
            for (int v = 0; v < 8; ++v)
                bfr.u[v] = ok ? bfr.u[v] : 0u;             // one cndmask_b32 each
            acc = wmma_bf16(af.bf, bfr.bf, acc);
        }
    }

    float* yb = y + (size_t)b * Cc * HW + y0 * Ww + x0 + n;
#pragma unroll
    for (int v = 0; v < 8; ++v) {
        const int m = coT * 16 + v + 8 * h;
        yb[(size_t)m * HW] = acc[v];
    }
}

// ============================================================================
// 2) Per-patch 4x4 DCT-II / inverse DCT (image layout, one thread per patch)
// ============================================================================
__global__ void __launch_bounds__(256)
dct2_patch(const float* __restrict__ in, float* __restrict__ out, int inverse)
{
    const int t  = blockIdx.x * 256 + threadIdx.x;   // 0..786431
    const int px = t & 31;
    const int py = (t >> 5) & 31;
    const int bc = t >> 10;                          // 0..767

    float M[16];
    if (inverse) {
#pragma unroll
        for (int nn = 0; nn < 4; ++nn)
#pragma unroll
            for (int k = 0; k < 4; ++k)
                M[nn * 4 + k] = c_D[k * 4 + nn] * (k == 0 ? 0.0625f : 0.125f);
    } else {
#pragma unroll
        for (int i = 0; i < 16; ++i) M[i] = c_D[i];
    }

    const float* p = in + (size_t)bc * HW + (py * 4) * Ww + px * 4;
    float X[4][4];
#pragma unroll
    for (int i = 0; i < 4; ++i) {
        float4 r = *reinterpret_cast<const float4*>(p + i * Ww);
        X[i][0] = r.x; X[i][1] = r.y; X[i][2] = r.z; X[i][3] = r.w;
    }
    float T[4][4];
#pragma unroll
    for (int i = 0; i < 4; ++i)
#pragma unroll
        for (int j = 0; j < 4; ++j) {
            float s = 0.f;
#pragma unroll
            for (int a = 0; a < 4; ++a) s += M[i * 4 + a] * X[a][j];
            T[i][j] = s;
        }
    float* o = out + (size_t)bc * HW + (py * 4) * Ww + px * 4;
#pragma unroll
    for (int i = 0; i < 4; ++i) {
        float4 r;
        float s0 = 0, s1 = 0, s2 = 0, s3 = 0;
#pragma unroll
        for (int bb = 0; bb < 4; ++bb) {
            s0 += T[i][bb] * M[0 * 4 + bb];
            s1 += T[i][bb] * M[1 * 4 + bb];
            s2 += T[i][bb] * M[2 * 4 + bb];
            s3 += T[i][bb] * M[3 * 4 + bb];
        }
        r.x = s0; r.y = s1; r.z = s2; r.w = s3;
        *reinterpret_cast<float4*>(o + i * Ww) = r;
    }
}

// ============================================================================
// 3) Per-patch F1 = fftQ @ fftK (4x4 matmul, one thread per patch)
// ============================================================================
__global__ void __launch_bounds__(256)
patch_mm(const float* __restrict__ q, const float* __restrict__ k,
         float* __restrict__ f1)
{
    const int t  = blockIdx.x * 256 + threadIdx.x;
    const int px = t & 31;
    const int py = (t >> 5) & 31;
    const int bc = t >> 10;
    const size_t base = (size_t)bc * HW + (py * 4) * Ww + px * 4;

    float Q[4][4], K[4][4];
#pragma unroll
    for (int i = 0; i < 4; ++i) {
        float4 rq = *reinterpret_cast<const float4*>(q + base + i * Ww);
        float4 rk = *reinterpret_cast<const float4*>(k + base + i * Ww);
        Q[i][0] = rq.x; Q[i][1] = rq.y; Q[i][2] = rq.z; Q[i][3] = rq.w;
        K[i][0] = rk.x; K[i][1] = rk.y; K[i][2] = rk.z; K[i][3] = rk.w;
    }
#pragma unroll
    for (int i = 0; i < 4; ++i) {
        float4 r;
        float s[4] = {0, 0, 0, 0};
#pragma unroll
        for (int a = 0; a < 4; ++a)
#pragma unroll
            for (int j = 0; j < 4; ++j) s[j] += Q[i][a] * K[a][j];
        r.x = s[0]; r.y = s[1]; r.z = s[2]; r.w = s[3];
        *reinterpret_cast<float4*>(f1 + base + i * Ww) = r;
    }
}

// ============================================================================
// 4) attn[b,i,j] = sum_l Fq[b,i,l]*Fk[b,j,l] — split-K(16) WMMA GEMM with
//    f32 global atomics into a zeroed buffer. Temperature applied in softmax.
// ============================================================================
__global__ void __launch_bounds__(256)
attn_gemm(const float* __restrict__ fq, const float* __restrict__ fk,
          float* __restrict__ attn)
{
    const int lane = threadIdx.x & 31;
    const int wid  = blockIdx.x * 8 + (threadIdx.x >> 5);   // 0..4607
    const int kc   = wid & 15;
    const int r    = wid >> 4;                              // 0..287
    const int b    = r / 36;
    const int t    = r - b * 36;
    const int iT   = t / 6, jT = t - (t / 6) * 6;

    const int m15 = lane & 15;
    const int h   = lane >> 4;
    const float* Arow = fq + (size_t)b * Cc * HW + (size_t)(iT * 16 + m15) * HW;
    const float* Brow = fk + (size_t)b * Cc * HW + (size_t)(jT * 16 + m15) * HW;

    v8f acc = {};
    const int l0beg = kc * 1024;
#pragma unroll 2
    for (int l0 = l0beg; l0 < l0beg + 1024; l0 += 32) {
        v16bf af, bf;
#pragma unroll
        for (int v = 0; v < 8; ++v) {
            float2 pa = *reinterpret_cast<const float2*>(Arow + l0 + a_kb(v, h));
            af[2 * v] = (__bf16)pa.x; af[2 * v + 1] = (__bf16)pa.y;
            float2 pb = *reinterpret_cast<const float2*>(Brow + l0 + b_kb(v, h));
            bf[2 * v] = (__bf16)pb.x; bf[2 * v + 1] = (__bf16)pb.y;
        }
        acc = wmma_bf16(af, bf, acc);
    }
#pragma unroll
    for (int v = 0; v < 8; ++v) {
        const int m = iT * 16 + v + 8 * h;
        atomicAdd(&attn[(size_t)b * Cc * Cc + m * Cc + jT * 16 + m15], acc[v]);
    }
}

// ============================================================================
// 5) softmax over rows of 96 (one wave per row), temperature scale fused.
// ============================================================================
__global__ void __launch_bounds__(256)
softmax96(float* __restrict__ attn, const float* __restrict__ temp)
{
    const int lane = threadIdx.x & 31;
    const int row  = blockIdx.x * 8 + (threadIdx.x >> 5);   // 0..767
    float* rp = attn + (size_t)row * Cc;
    const float ts = temp[0];
    float v0 = rp[lane] * ts, v1 = rp[lane + 32] * ts, v2 = rp[lane + 64] * ts;
    float mx = fmaxf(v0, fmaxf(v1, v2));
#pragma unroll
    for (int off = 16; off > 0; off >>= 1) mx = fmaxf(mx, __shfl_xor(mx, off, 32));
    float e0 = __expf(v0 - mx), e1 = __expf(v1 - mx), e2 = __expf(v2 - mx);
    float s = e0 + e1 + e2;
#pragma unroll
    for (int off = 16; off > 0; off >>= 1) s += __shfl_xor(s, off, 32);
    const float inv = 1.0f / s;
    rp[lane] = e0 * inv; rp[lane + 32] = e1 * inv; rp[lane + 64] = e2 * inv;
}

// ============================================================================
// 6) Fcfr[b,i,l] = sum_j attn[b,i,j] * F1[b,j,l]  (M=96, K=96, N=16384)
// ============================================================================
__global__ void __launch_bounds__(256)
fcfr_gemm(const float* __restrict__ attn, const float* __restrict__ f1,
          float* __restrict__ out)
{
    const int lane = threadIdx.x & 31;
    const int wid  = blockIdx.x * 8 + (threadIdx.x >> 5);   // 0..49151
    const int b    = wid / 6144;
    const int t    = wid - b * 6144;
    const int mT   = t >> 10;          // 0..5
    const int nT   = t & 1023;         // 0..1023

    const int m15 = lane & 15;
    const int h   = lane >> 4;
    const float* Arow = attn + (size_t)b * Cc * Cc + (size_t)(mT * 16 + m15) * Cc;
    const float* Bcol = f1 + (size_t)b * Cc * HW + nT * 16 + m15;

    v8f acc = {};
#pragma unroll
    for (int k0 = 0; k0 < 96; k0 += 32) {
        v16bf af, bf;
#pragma unroll
        for (int v = 0; v < 8; ++v) {
            float2 pa = *reinterpret_cast<const float2*>(Arow + k0 + a_kb(v, h));
            af[2 * v] = (__bf16)pa.x; af[2 * v + 1] = (__bf16)pa.y;
            const int k = k0 + b_kb(v, h);
            bf[2 * v]     = (__bf16)Bcol[(size_t)k * HW];
            bf[2 * v + 1] = (__bf16)Bcol[(size_t)(k + 1) * HW];
        }
        acc = wmma_bf16(af, bf, acc);
    }
#pragma unroll
    for (int v = 0; v < 8; ++v) {
        const int m = mT * 16 + v + 8 * h;
        out[(size_t)b * Cc * HW + (size_t)m * HW + nT * 16 + m15] = acc[v];
    }
}

// ============================================================================
// 7) Final fused batched GEMM over 768 (b,c) pairs, 128x128x128:
//    out = Fcfrout@F_Q + F_V - 0.5 * (Fcfrout@F_V)
// ============================================================================
__global__ void __launch_bounds__(256)
final_gemm(const float* __restrict__ fc, const float* __restrict__ fq,
           const float* __restrict__ fv, float* __restrict__ out)
{
    const int lane = threadIdx.x & 31;
    const int wid  = blockIdx.x * 8 + (threadIdx.x >> 5);   // 0..49151
    const int bc   = wid >> 6;           // 0..767
    const int t    = wid & 63;
    const int mT   = t >> 3, nT = t & 7;

    const int m15 = lane & 15;
    const int h   = lane >> 4;
    const size_t base = (size_t)bc * HW;
    const float* Arow = fc + base + (size_t)(mT * 16 + m15) * Ww;
    const float* Bq   = fq + base + nT * 16 + m15;
    const float* Bv   = fv + base + nT * 16 + m15;

    v8f accQ = {}, accV = {};
#pragma unroll
    for (int k0 = 0; k0 < 128; k0 += 32) {
        v16bf af, bq, bv;
#pragma unroll
        for (int v = 0; v < 8; ++v) {
            float2 pa = *reinterpret_cast<const float2*>(Arow + k0 + a_kb(v, h));
            af[2 * v] = (__bf16)pa.x; af[2 * v + 1] = (__bf16)pa.y;
            const int k = k0 + b_kb(v, h);
            bq[2 * v]     = (__bf16)Bq[(size_t)k * Ww];
            bq[2 * v + 1] = (__bf16)Bq[(size_t)(k + 1) * Ww];
            bv[2 * v]     = (__bf16)Bv[(size_t)k * Ww];
            bv[2 * v + 1] = (__bf16)Bv[(size_t)(k + 1) * Ww];
        }
        accQ = wmma_bf16(af, bq, accQ);
        accV = wmma_bf16(af, bv, accV);
    }
#pragma unroll
    for (int v = 0; v < 8; ++v) {
        const int m = mT * 16 + v + 8 * h;
        const int nn = nT * 16 + m15;
        const float fvv = fv[base + (size_t)m * Ww + nn];
        out[base + (size_t)m * Ww + nn] = accQ[v] + fvv - 0.5f * accV[v];
    }
}

// ============================================================================
extern "C" void kernel_launch(void* const* d_in, const int* in_sizes, int n_in,
                              void* d_out, int out_size, void* d_ws, size_t ws_size,
                              hipStream_t stream)
{
    const float* in1  = (const float*)d_in[0];
    const float* in2  = (const float*)d_in[1];
    const float* wq   = (const float*)d_in[2];
    const float* bq   = (const float*)d_in[3];
    const float* wk   = (const float*)d_in[4];
    const float* bk   = (const float*)d_in[5];
    const float* wv   = (const float*)d_in[6];
    const float* bv   = (const float*)d_in[7];
    const float* temp = (const float*)d_in[8];
    float* out = (float*)d_out;

    float* ws   = (float*)d_ws;
    float* fQ   = ws;                 // conv Q output (kept for final GEMM)
    float* fK   = ws + TS;            // conv K output -> fftK in place
    float* fV   = ws + 2 * TS;        // conv V output (kept for final GEMM)
    float* fftQ = ws + 3 * TS;        // DCT(F_Q); reused as Fcfr/Fcfrout later
    float* F1   = ws + 4 * TS;        // per-patch fftQ@fftK
    float* attn = ws + 5 * TS;        // [8,96,96]

    // 1) three convolutions (implicit GEMM, bf16 WMMA, f32 accumulate)
    conv3x3_wmma<<<6144, 256, 0, stream>>>(in1, wq, bq, fQ);
    conv3x3_wmma<<<6144, 256, 0, stream>>>(in1, wk, bk, fK);
    conv3x3_wmma<<<6144, 256, 0, stream>>>(in2, wv, bv, fV);
    // 2) per-patch DCT-II
    dct2_patch<<<3072, 256, 0, stream>>>(fQ, fftQ, 0);
    dct2_patch<<<3072, 256, 0, stream>>>(fK, fK, 0);      // in place
    // 3) per-patch 4x4 matmul
    patch_mm<<<3072, 256, 0, stream>>>(fftQ, fK, F1);
    // 4) channel-channel attention GEMM (split-K, atomic f32 accumulate)
    hipMemsetAsync((void*)attn, 0, (size_t)Bn * Cc * Cc * sizeof(float), stream);
    attn_gemm<<<576, 256, 0, stream>>>(fftQ, fK, attn);
    // 5) softmax (temperature fused)
    softmax96<<<96, 256, 0, stream>>>(attn, temp);
    // 6) Fcfr = attn @ F1   (overwrites fftQ buffer, no longer needed)
    fcfr_gemm<<<6144, 256, 0, stream>>>(attn, F1, fftQ);
    // 7) inverse DCT in place -> Fcfrout
    dct2_patch<<<3072, 256, 0, stream>>>(fftQ, fftQ, 1);
    // 8) fused final batched GEMMs + elementwise
    final_gemm<<<6144, 256, 0, stream>>>(fftQ, fQ, fV, out);
}
